// LATTE_85083302134347
// MI455X (gfx1250) — matrix operand order repaired and checked by reading, not compiled, not run on hardware
//
#include <hip/hip_runtime.h>
#include <hip/hip_bf16.h>
#include <math.h>

typedef _Float16 h8   __attribute__((ext_vector_type(8)));
typedef _Float16 v16h __attribute__((ext_vector_type(16)));
typedef float    v8f  __attribute__((ext_vector_type(8)));

// ---------------------------------------------------------------------------
// Generic WMMA GEMM: C[Mpad x Npad] = act( A[Mpad x K] * B[Npad x K]^T + bias )
// A: f16 row-major (ld = K), B: f16 "torch layout" (N x K, ld = K).
// One wave (32 threads) computes one 16x16 C tile. K multiple of 32.
// act: 0 = none, 1 = relu, 2 = sigmoid.
// ---------------------------------------------------------------------------
__global__ void gemm_wmma_kernel(const _Float16* __restrict__ A,
                                 const _Float16* __restrict__ B,
                                 float* __restrict__ C,
                                 int K, int ldc, int nValid,
                                 const float* __restrict__ bias0,
                                 const float* __restrict__ bias1,
                                 int act)
{
    const int lane = threadIdx.x;      // 0..31
    const int l16  = lane & 15;
    const int hsel = lane >> 4;        // 0 or 1

    const _Float16* Ap = A + (size_t)(blockIdx.x * 16 + l16) * K;  // row m
    const _Float16* Bp = B + (size_t)(blockIdx.y * 16 + l16) * K;  // row n

    v8f acc = {0.f, 0.f, 0.f, 0.f, 0.f, 0.f, 0.f, 0.f};

    for (int k = 0; k < K; k += 32) {
        // A 16x32 fragment: lane holds k = hsel*8 + [0..7] and 16 + hsel*8 + [0..7]
        h8 a0 = *(const h8*)(Ap + k + hsel * 8);
        h8 a1 = *(const h8*)(Ap + k + 16 + hsel * 8);
        // B 32x16 fragment: lane holds k = hsel*8 + [0..7] and 16 + hsel*8 + [0..7] at fixed n
        h8 b0 = *(const h8*)(Bp + k + hsel * 8);
        h8 b1 = *(const h8*)(Bp + k + 16 + hsel * 8);
        v16h av, bv;
#pragma unroll
        for (int e = 0; e < 8; ++e) {
            av[e] = a0[e]; av[e + 8] = a1[e];
            bv[e] = b0[e]; bv[e + 8] = b1[e];
        }
        acc = __builtin_amdgcn_wmma_f32_16x16x32_f16(
            false, av, false, bv, (short)0, acc, false, false);
    }

    const int n = blockIdx.y * 16 + l16;
    float bias = 0.f;
    if (n < nValid) {
        if (bias0) bias += bias0[n];
        if (bias1) bias += bias1[n];
    }
#pragma unroll
    for (int v = 0; v < 8; ++v) {
        int m = blockIdx.x * 16 + v + 8 * hsel;
        float x = acc[v] + bias;
        if (act == 1)      x = fmaxf(x, 0.f);
        else if (act == 2) x = 1.f / (1.f + __expf(-x));
        C[(size_t)m * ldc + n] = x;
    }
}

// ---------------------------------------------------------------------------
__global__ void zero_u32(unsigned int* __restrict__ p, size_t n)
{
    size_t i = (size_t)blockIdx.x * blockDim.x + threadIdx.x;
    size_t stride = (size_t)gridDim.x * blockDim.x;
    for (; i < n; i += stride) p[i] = 0u;
}

// Pack f32 weight (N x K) into zero-padded f16 (Npad x Kpad), torch layout kept.
__global__ void pack_weight(const float* __restrict__ W, _Float16* __restrict__ O,
                            int N, int K, int Kpad, size_t total)
{
    size_t i = (size_t)blockIdx.x * blockDim.x + threadIdx.x;
    size_t stride = (size_t)gridDim.x * blockDim.x;
    for (; i < total; i += stride) {
        int n = (int)(i / Kpad), k = (int)(i % Kpad);
        float v = (n < N && k < K) ? W[(size_t)n * K + k] : 0.f;
        O[i] = (_Float16)v;
    }
}

// Copy valid region of an f32 activation into a padded f16 buffer.
__global__ void pack_act(const float* __restrict__ X, int ldx,
                         _Float16* __restrict__ O, int ldo, int rows, int cols)
{
    size_t total = (size_t)rows * cols;
    size_t i = (size_t)blockIdx.x * blockDim.x + threadIdx.x;
    size_t stride = (size_t)gridDim.x * blockDim.x;
    for (; i < total; i += stride) {
        int r = (int)(i / cols), c = (int)(i % cols);
        O[(size_t)r * ldo + c] = (_Float16)X[(size_t)r * ldx + c];
    }
}

// Uc = c.reshape(40, 2200) packed into f16 (48 x 2208)
__global__ void uc_pack(const float* __restrict__ LO, _Float16* __restrict__ O)
{
    size_t total = (size_t)40 * 2200;
    size_t i = (size_t)blockIdx.x * blockDim.x + threadIdx.x;
    size_t stride = (size_t)gridDim.x * blockDim.x;
    for (; i < total; i += stride) {
        int b = (int)(i / 2200), f = (int)(i % 2200);
        O[(size_t)b * 2208 + f] =
            (_Float16)LO[((size_t)b * 11 + f / 200) * 200 + (f % 200)];
    }
}

// ---------------------------------------------------------------------------
// Char CNN: per (b,s) row -> gather emb (8x16), conv width 5 over 12 positions,
// relu + maxpool over time. Writes cols [0,100) of HX row.
// ---------------------------------------------------------------------------
__global__ void char_layer_kernel(const int* __restrict__ chars, int row_offset,
                                  const float* __restrict__ cemb,
                                  const float* __restrict__ convw,
                                  const float* __restrict__ convb,
                                  float* __restrict__ HX)
{
    __shared__ float e[8][16];
    const int r = blockIdx.x, tid = threadIdx.x;     // 128 threads
    {
        int w = tid >> 3, cd = tid & 7;
        int ci = chars[(size_t)r * 16 + w];
        e[cd][w] = cemb[(size_t)ci * 8 + cd];
    }
    __syncthreads();
    if (tid < 100) {
        const int cc = tid;
        float wreg[40];
#pragma unroll
        for (int cd = 0; cd < 8; ++cd)
#pragma unroll
            for (int k = 0; k < 5; ++k)
                wreg[cd * 5 + k] = convw[((size_t)cc * 8 + cd) * 5 + k];
        const float bb = convb[cc];
        float m = -1e30f;
        for (int t = 0; t < 12; ++t) {
            float s = bb;
#pragma unroll
            for (int cd = 0; cd < 8; ++cd)
#pragma unroll
                for (int k = 0; k < 5; ++k)
                    s += e[cd][t + k] * wreg[cd * 5 + k];
            m = fmaxf(m, s);
        }
        HX[(size_t)(row_offset + r) * 200 + cc] = fmaxf(m, 0.f);
    }
}

__global__ void word_gather_kernel(const int* __restrict__ idx, int row_offset,
                                   const float* __restrict__ wemb,
                                   float* __restrict__ HX)
{
    const int r = blockIdx.x, tid = threadIdx.x;     // 128 threads
    if (tid < 100)
        HX[(size_t)(row_offset + r) * 200 + 100 + tid] =
            wemb[(size_t)idx[r] * 100 + tid];
}

// x = g*h + (1-g)*x ; writes both f32 (for next layer read) and f16 (for GEMM)
__global__ void highway_combine(float* __restrict__ HX,
                                const float* __restrict__ Y,
                                const float* __restrict__ G,
                                _Float16* __restrict__ HX16)
{
    size_t total = (size_t)520 * 200;
    size_t i = (size_t)blockIdx.x * blockDim.x + threadIdx.x;
    size_t stride = (size_t)gridDim.x * blockDim.x;
    for (; i < total; i += stride) {
        int r = (int)(i / 200), c = (int)(i % 200);
        float x = HX[i];
        float g = G[(size_t)r * 208 + c];
        float h = Y[(size_t)r * 208 + c];
        float nx = g * h + (1.f - g) * x;
        HX[i] = nx;
        HX16[(size_t)r * 224 + c] = (_Float16)nx;
    }
}

// ---------------------------------------------------------------------------
// LSTM recurrence (one direction). XG = x@Wih^T + bih + bhh precomputed.
// One block per batch element; h/c/g in LDS. torch gate order i,f,g,o.
// ---------------------------------------------------------------------------
__global__ void lstm_kernel(const float* __restrict__ XG,
                            const float* __restrict__ whh,
                            float* __restrict__ OUT,
                            int rowbase, int T, int reverse, int outoff)
{
    __shared__ float h[100], cs[100], g[400];
    const int b = blockIdx.x, tid = threadIdx.x;     // 128 threads
    for (int j = tid; j < 100; j += 128) { h[j] = 0.f; cs[j] = 0.f; }
    __syncthreads();
    for (int step = 0; step < T; ++step) {
        int t = reverse ? (T - 1 - step) : step;
        size_t row = (size_t)rowbase + (size_t)b * T + t;
        for (int j = tid; j < 400; j += 128) {
            float s = XG[row * 400 + j];
            const float* wr = whh + (size_t)j * 100;
            for (int k = 0; k < 100; ++k) s += h[k] * wr[k];
            g[j] = s;
        }
        __syncthreads();
        if (tid < 100) {
            float ig = g[tid], fg = g[100 + tid], gg = g[200 + tid], og = g[300 + tid];
            float c2 = (1.f / (1.f + __expf(-fg))) * cs[tid] +
                       (1.f / (1.f + __expf(-ig))) * tanhf(gg);
            float h2 = (1.f / (1.f + __expf(-og))) * tanhf(c2);
            cs[tid] = c2; h[tid] = h2;
            OUT[row * 200 + outoff + tid] = h2;
        }
        __syncthreads();
    }
}

// ---------------------------------------------------------------------------
// Attention flow with QL=1: softmax over QL is trivial (c2q = q).
// Writes the fc1 A-matrix directly as f16 (48 x 8800, rows 40..47 pre-zeroed).
// ---------------------------------------------------------------------------
__global__ void att_kernel(const float* __restrict__ LO, int qbase,
                           const float* __restrict__ wc, const float* __restrict__ bc,
                           const float* __restrict__ wq, const float* __restrict__ bq,
                           const float* __restrict__ wcq, const float* __restrict__ bcq,
                           _Float16* __restrict__ Apack)
{
    __shared__ float q[200], sv[11], bvec[11], q2c[200], red[256];
    __shared__ float sq_sh;
    const int b = blockIdx.x, tid = threadIdx.x;     // 256 threads
    if (tid < 200) q[tid] = LO[(size_t)(qbase + b) * 200 + tid];
    __syncthreads();

    float part = (tid < 200) ? q[tid] * wq[tid] : 0.f;
    red[tid] = part; __syncthreads();
    for (int s = 128; s > 0; s >>= 1) { if (tid < s) red[tid] += red[tid + s]; __syncthreads(); }
    if (tid == 0) sq_sh = red[0] + bq[0];
    __syncthreads();

    for (int i = 0; i < 11; ++i) {
        const float* ci = LO + ((size_t)b * 11 + i) * 200;
        float p = 0.f;
        if (tid < 200) { float cv = ci[tid]; p = cv * wc[tid] + cv * q[tid] * wcq[tid]; }
        red[tid] = p; __syncthreads();
        for (int s = 128; s > 0; s >>= 1) { if (tid < s) red[tid] += red[tid + s]; __syncthreads(); }
        if (tid == 0) sv[i] = red[0] + bc[0] + sq_sh + bcq[0];
        __syncthreads();
    }
    if (tid == 0) {
        float mx = -1e30f;
        for (int i = 0; i < 11; ++i) mx = fmaxf(mx, sv[i]);
        float sum = 0.f;
        for (int i = 0; i < 11; ++i) { bvec[i] = __expf(sv[i] - mx); sum += bvec[i]; }
        for (int i = 0; i < 11; ++i) bvec[i] /= sum;
    }
    __syncthreads();
    if (tid < 200) {
        float acc = 0.f;
        for (int i = 0; i < 11; ++i) acc += bvec[i] * LO[((size_t)b * 11 + i) * 200 + tid];
        q2c[tid] = acc;
    }
    __syncthreads();
    for (int i = 0; i < 11; ++i) {
        const float* ci = LO + ((size_t)b * 11 + i) * 200;
        for (int d = tid; d < 800; d += 256) {
            float v;
            if (d < 200)      v = ci[d];
            else if (d < 400) v = q[d - 200];
            else if (d < 600) v = ci[d - 400] * q[d - 400];
            else              v = ci[d - 600] * q2c[d - 600];
            Apack[(size_t)b * 8800 + i * 800 + d] = (_Float16)v;
        }
    }
}

// ---------------------------------------------------------------------------
__global__ void softmax_rows(const float* __restrict__ X, float* __restrict__ Y,
                             int width)
{
    __shared__ float red[256];
    __shared__ float stat;
    const int b = blockIdx.x, tid = threadIdx.x;     // 256 threads
    const float* x = X + (size_t)b * width;
    float mx = -1e30f;
    for (int k = tid; k < width; k += 256) mx = fmaxf(mx, x[k]);
    red[tid] = mx; __syncthreads();
    for (int s = 128; s > 0; s >>= 1) { if (tid < s) red[tid] = fmaxf(red[tid], red[tid + s]); __syncthreads(); }
    if (tid == 0) stat = red[0];
    __syncthreads();
    const float m = stat;
    float sum = 0.f;
    for (int k = tid; k < width; k += 256) sum += __expf(x[k] - m);
    __syncthreads();
    red[tid] = sum; __syncthreads();
    for (int s = 128; s > 0; s >>= 1) { if (tid < s) red[tid] += red[tid + s]; __syncthreads(); }
    if (tid == 0) stat = red[0];
    __syncthreads();
    const float inv = 1.f / stat;
    float* y = Y + (size_t)b * width;
    for (int k = tid; k < width; k += 256) y[k] = __expf(x[k] - m) * inv;
}

// per-column cosine combine over batch axis
__global__ void latent_cols(const float* __restrict__ P, const float* __restrict__ Cm,
                            float* __restrict__ gl)
{
    int j = blockIdx.x * blockDim.x + threadIdx.x;
    if (j >= 2048) return;
    float num = 0.f, sp = 0.f, sc = 0.f;
    for (int b = 0; b < 40; ++b) {
        float p = P[(size_t)b * 2048 + j], c = Cm[(size_t)b * 2048 + j];
        num += p * c; sp += p * p; sc += c * c;
    }
    float den = fmaxf(sqrtf(sp), 1e-6f) * fmaxf(sqrtf(sc), 1e-6f);
    gl[j] = num / den;
}

__global__ void glg_kernel(const float* __restrict__ gl, const float* __restrict__ gw,
                           const float* __restrict__ gwb, float* __restrict__ out)
{
    __shared__ float red[256];
    const int tid = threadIdx.x;
    float s = 0.f;
    for (int k = tid; k < 2048; k += 256) s += gl[k] * gw[k];
    red[tid] = s; __syncthreads();
    for (int st = 128; st > 0; st >>= 1) { if (tid < st) red[tid] += red[tid + st]; __syncthreads(); }
    if (tid == 0) out[0] = red[0] + gwb[0];
}

// y = relu(V @ W^T + b), N=3; one wave per output feature
__global__ void fc56_kernel(const float* __restrict__ V, const float* __restrict__ W,
                            const float* __restrict__ bias, float* __restrict__ out1,
                            float* __restrict__ out2)
{
    __shared__ float red[96];
    const int b = blockIdx.x, tid = threadIdx.x;     // 96 threads
    const int j = tid >> 5, lane = tid & 31;
    float s = 0.f;
    const float* v = V + (size_t)b * 2048;
    const float* w = W + (size_t)j * 2048;
    for (int k = lane; k < 2048; k += 32) s += v[k] * w[k];
    red[tid] = s; __syncthreads();
    if (lane == 0) {
        float t = 0.f;
        for (int l = 0; l < 32; ++l) t += red[j * 32 + l];
        float y = fmaxf(t + bias[j], 0.f);
        out1[b * 3 + j] = y;
        if (out2) out2[b * 3 + j] = y;
    }
}

__global__ void final_r_kernel(const float* __restrict__ F, const float* __restrict__ fw,
                               const float* __restrict__ fwb, const float* __restrict__ glg,
                               float* __restrict__ out)
{
    __shared__ float red[256];
    const int b = blockIdx.x, tid = threadIdx.x;     // 256 threads
    float s = (tid < 200) ? F[(size_t)b * 208 + tid] * fw[tid] : 0.f;
    red[tid] = s; __syncthreads();
    for (int st = 128; st > 0; st >>= 1) { if (tid < st) red[tid] += red[tid + st]; __syncthreads(); }
    if (tid == 0) out[b] = red[0] + fwb[0] + glg[0];
}

// ---------------------------------------------------------------------------
extern "C" void kernel_launch(void* const* d_in, const int* in_sizes, int n_in,
                              void* d_out, int out_size, void* d_ws, size_t ws_size,
                              hipStream_t stream)
{
    (void)in_sizes; (void)n_in; (void)out_size; (void)ws_size;

    const int*   c_char    = (const int*)  d_in[0];
    const int*   p_char    = (const int*)  d_in[1];
    const int*   n_char    = (const int*)  d_in[2];
    const int*   c_word    = (const int*)  d_in[3];
    const int*   p_word    = (const int*)  d_in[4];
    const int*   n_word    = (const int*)  d_in[5];
    const float* word_emb  = (const float*)d_in[6];
    const float* char_emb  = (const float*)d_in[7];
    const float* conv_w    = (const float*)d_in[8];
    const float* conv_b    = (const float*)d_in[9];
    const float* hw_lin_w  = (const float*)d_in[10];
    const float* hw_lin_b  = (const float*)d_in[11];
    const float* hw_gate_w = (const float*)d_in[12];
    const float* hw_gate_b = (const float*)d_in[13];
    const float* wih_f     = (const float*)d_in[14];
    const float* whh_f     = (const float*)d_in[15];
    const float* bih_f     = (const float*)d_in[16];
    const float* bhh_f     = (const float*)d_in[17];
    const float* wih_b     = (const float*)d_in[18];
    const float* whh_b     = (const float*)d_in[19];
    const float* bih_b     = (const float*)d_in[20];
    const float* bhh_b     = (const float*)d_in[21];
    const float* att_wc    = (const float*)d_in[22];
    const float* att_bc    = (const float*)d_in[23];
    const float* att_wq    = (const float*)d_in[24];
    const float* att_bq    = (const float*)d_in[25];
    const float* att_wcq   = (const float*)d_in[26];
    const float* att_bcq   = (const float*)d_in[27];
    const float* fc1_w     = (const float*)d_in[28];
    const float* fc1_b     = (const float*)d_in[29];
    const float* fc3_w     = (const float*)d_in[30];
    const float* fc3_b     = (const float*)d_in[31];
    const float* fc4_w     = (const float*)d_in[32];
    const float* fc4_b     = (const float*)d_in[33];
    const float* fc5_w     = (const float*)d_in[34];
    const float* fc5_b     = (const float*)d_in[35];
    const float* fc6_w     = (const float*)d_in[36];
    const float* fc6_b     = (const float*)d_in[37];
    const float* fw_w      = (const float*)d_in[38];
    const float* fw_b      = (const float*)d_in[39];
    const float* gw_w      = (const float*)d_in[40];
    const float* gw_b      = (const float*)d_in[41];
    float* out = (float*)d_out;

    char* ws = (char*)d_ws;
    size_t off = 0;
    auto alloc = [&](size_t bytes) -> void* {
        void* p = ws + off;
        off += (bytes + 255) & ~(size_t)255;
        return p;
    };

    // f32 intermediates
    float* HX    = (float*)alloc((size_t)528 * 200 * 4);   // [char | word] then highway
    float* Ylin  = (float*)alloc((size_t)528 * 208 * 4);
    float* Ygate = (float*)alloc((size_t)528 * 208 * 4);
    float* XGf   = (float*)alloc((size_t)528 * 400 * 4);
    float* XGb   = (float*)alloc((size_t)528 * 400 * 4);
    float* LO    = (float*)alloc((size_t)520 * 200 * 4);   // BiLSTM out [fwd|bwd]
    float* Fp    = (float*)alloc((size_t)48 * 208 * 4);
    float* Fn    = (float*)alloc((size_t)48 * 208 * 4);
    float* Vp_p  = (float*)alloc((size_t)48 * 2048 * 4);
    float* Vp_n  = (float*)alloc((size_t)48 * 2048 * 4);
    float* Vc    = (float*)alloc((size_t)48 * 2048 * 4);   // shared by p/n branches
    float* PH_p  = (float*)alloc((size_t)40 * 2048 * 4);
    float* PH_n  = (float*)alloc((size_t)40 * 2048 * 4);
    float* CH    = (float*)alloc((size_t)40 * 2048 * 4);
    float* gl_p  = (float*)alloc((size_t)2048 * 4);
    float* gl_n  = (float*)alloc((size_t)2048 * 4);
    float* glg   = (float*)alloc((size_t)2 * 4);

    // f16 activation buffers (zero-padded; zeroed every call)
    char* act_begin = ws + off;
    _Float16* HX16 = (_Float16*)alloc((size_t)528 * 224 * 2);
    _Float16* Ap_p = (_Float16*)alloc((size_t)48 * 8800 * 2);
    _Float16* Ap_n = (_Float16*)alloc((size_t)48 * 8800 * 2);
    _Float16* Uc16 = (_Float16*)alloc((size_t)48 * 2208 * 2);
    _Float16* Up_p = (_Float16*)alloc((size_t)48 * 224 * 2);
    _Float16* Up_n = (_Float16*)alloc((size_t)48 * 224 * 2);
    char* act_end = ws + off;

    // f16 weight packs
    _Float16* Whl0 = (_Float16*)alloc((size_t)208 * 224 * 2);
    _Float16* Whl1 = (_Float16*)alloc((size_t)208 * 224 * 2);
    _Float16* Whg0 = (_Float16*)alloc((size_t)208 * 224 * 2);
    _Float16* Whg1 = (_Float16*)alloc((size_t)208 * 224 * 2);
    _Float16* Wif  = (_Float16*)alloc((size_t)400 * 224 * 2);
    _Float16* Wib  = (_Float16*)alloc((size_t)400 * 224 * 2);
    _Float16* W1   = (_Float16*)alloc((size_t)208 * 8800 * 2);
    _Float16* W3   = (_Float16*)alloc((size_t)2048 * 224 * 2);
    _Float16* W4   = (_Float16*)alloc((size_t)2048 * 2208 * 2);

    // --- zero padded f16 activation region ---
    zero_u32<<<512, 256, 0, stream>>>((unsigned int*)act_begin,
                                      (size_t)(act_end - act_begin) / 4);

    // --- weight packing ---
    auto packw = [&](const float* W, _Float16* O, int N, int K, int Npad, int Kpad) {
        size_t total = (size_t)Npad * Kpad;
        int blocks = (int)((total + 255) / 256);
        pack_weight<<<blocks, 256, 0, stream>>>(W, O, N, K, Kpad, total);
    };
    packw(hw_lin_w + 0 * 40000,  Whl0, 200, 200, 208, 224);
    packw(hw_lin_w + 1 * 40000,  Whl1, 200, 200, 208, 224);
    packw(hw_gate_w + 0 * 40000, Whg0, 200, 200, 208, 224);
    packw(hw_gate_w + 1 * 40000, Whg1, 200, 200, 208, 224);
    packw(wih_f, Wif, 400, 200, 400, 224);
    packw(wih_b, Wib, 400, 200, 400, 224);
    packw(fc1_w, W1, 200, 8800, 208, 8800);
    packw(fc3_w, W3, 2048, 200, 2048, 224);
    packw(fc4_w, W4, 2048, 2200, 2048, 2208);

    // --- char CNN + word embedding gather into HX (rows: c 0..439, p 440..479, n 480..519) ---
    char_layer_kernel<<<440, 128, 0, stream>>>(c_char, 0,   char_emb, conv_w, conv_b, HX);
    char_layer_kernel<<<40,  128, 0, stream>>>(p_char, 440, char_emb, conv_w, conv_b, HX);
    char_layer_kernel<<<40,  128, 0, stream>>>(n_char, 480, char_emb, conv_w, conv_b, HX);
    word_gather_kernel<<<440, 128, 0, stream>>>(c_word, 0,   word_emb, HX);
    word_gather_kernel<<<40,  128, 0, stream>>>(p_word, 440, word_emb, HX);
    word_gather_kernel<<<40,  128, 0, stream>>>(n_word, 480, word_emb, HX);
    pack_act<<<407, 256, 0, stream>>>(HX, 200, HX16, 224, 520, 200);

    auto gemm = [&](const _Float16* A, const _Float16* Bw, float* C, int Mt, int Nt,
                    int K, int nValid, const float* b0, const float* b1, int act) {
        dim3 g(Mt, Nt);
        gemm_wmma_kernel<<<g, 32, 0, stream>>>(A, Bw, C, K, Nt * 16, nValid, b0, b1, act);
    };

    // --- highway (2 layers) ---
    for (int i = 0; i < 2; ++i) {
        gemm(HX16, i ? Whl1 : Whl0, Ylin,  33, 13, 224, 200, hw_lin_b + i * 200,  nullptr, 1);
        gemm(HX16, i ? Whg1 : Whg0, Ygate, 33, 13, 224, 200, hw_gate_b + i * 200, nullptr, 2);
        highway_combine<<<407, 256, 0, stream>>>(HX, Ylin, Ygate, HX16);
    }

    // --- LSTM input projections (WMMA) then recurrence (VALU) ---
    gemm(HX16, Wif, XGf, 33, 25, 224, 400, bih_f, bhh_f, 0);
    gemm(HX16, Wib, XGb, 33, 25, 224, 400, bih_b, bhh_b, 0);
    lstm_kernel<<<40, 128, 0, stream>>>(XGf, whh_f, LO, 0,   11, 0, 0);
    lstm_kernel<<<40, 128, 0, stream>>>(XGb, whh_b, LO, 0,   11, 1, 100);
    lstm_kernel<<<40, 128, 0, stream>>>(XGf, whh_f, LO, 440, 1,  0, 0);
    lstm_kernel<<<40, 128, 0, stream>>>(XGb, whh_b, LO, 440, 1,  1, 100);
    lstm_kernel<<<40, 128, 0, stream>>>(XGf, whh_f, LO, 480, 1,  0, 0);
    lstm_kernel<<<40, 128, 0, stream>>>(XGb, whh_b, LO, 480, 1,  1, 100);

    // --- attention flow (QL=1) -> fc1 A-matrices ---
    att_kernel<<<40, 256, 0, stream>>>(LO, 440, att_wc, att_bc, att_wq, att_bq,
                                       att_wcq, att_bcq, Ap_p);
    att_kernel<<<40, 256, 0, stream>>>(LO, 480, att_wc, att_bc, att_wq, att_bq,
                                       att_wcq, att_bcq, Ap_n);

    // --- pack Uc / Up and run the big GEMMs ---
    uc_pack<<<344, 256, 0, stream>>>(LO, Uc16);
    pack_act<<<32, 256, 0, stream>>>(LO + (size_t)440 * 200, 200, Up_p, 224, 40, 200);
    pack_act<<<32, 256, 0, stream>>>(LO + (size_t)480 * 200, 200, Up_n, 224, 40, 200);

    gemm(Ap_p, W1, Fp,   3, 13,  8800, 200,  fc1_b, nullptr, 1);
    gemm(Ap_n, W1, Fn,   3, 13,  8800, 200,  fc1_b, nullptr, 1);
    gemm(Up_p, W3, Vp_p, 3, 128, 224,  2048, fc3_b, nullptr, 0);
    gemm(Up_n, W3, Vp_n, 3, 128, 224,  2048, fc3_b, nullptr, 0);
    gemm(Uc16, W4, Vc,   3, 128, 2208, 2048, fc4_b, nullptr, 0);

    // --- latent cosine path ---
    softmax_rows<<<40, 256, 0, stream>>>(Vp_p, PH_p, 2048);
    softmax_rows<<<40, 256, 0, stream>>>(Vp_n, PH_n, 2048);
    softmax_rows<<<40, 256, 0, stream>>>(Vc,   CH,   2048);
    latent_cols<<<8, 256, 0, stream>>>(PH_p, CH, gl_p);
    latent_cols<<<8, 256, 0, stream>>>(PH_n, CH, gl_n);
    glg_kernel<<<1, 256, 0, stream>>>(gl_p, gw_w, gw_b, glg + 0);
    glg_kernel<<<1, 256, 0, stream>>>(gl_n, gw_w, gw_b, glg + 1);

    // --- heads: out = [r_p(40), r_n(40), yp_p(120), yc_p(120), yp_n(120), yc_n(120)] ---
    fc56_kernel<<<40, 96, 0, stream>>>(Vp_p, fc5_w, fc5_b, out + 80,  nullptr);
    fc56_kernel<<<40, 96, 0, stream>>>(Vp_n, fc5_w, fc5_b, out + 320, nullptr);
    fc56_kernel<<<40, 96, 0, stream>>>(Vc,   fc6_w, fc6_b, out + 200, out + 440);
    final_r_kernel<<<40, 256, 0, stream>>>(Fp, fw_w, fw_b, glg + 0, out + 0);
    final_r_kernel<<<40, 256, 0, stream>>>(Fn, fw_w, fw_b, glg + 1, out + 40);
}
